// Transformer_69836168233317
// MI455X (gfx1250) — matrix operand order repaired and checked by reading, not compiled, not run on hardware
//
#include <hip/hip_runtime.h>
#include <hip/hip_bf16.h>
#include <math.h>

// ---------------------------------------------------------------------------
// Model dims
// ---------------------------------------------------------------------------
#define BB    32
#define NN    201
#define DIMD  512
#define HWW   11
#define PP    121      // 11*11
#define MIDC  9
#define MLPD  1024
#define HEADS 8
#define DH    64
#define ROWS  6432     // B*N
#define NTK   13       // ceil(201/16)
#define NP208 208
#define NP224 224
#define SCALEF 0.125f  // 64^-0.5

typedef _Float16 v16h __attribute__((ext_vector_type(16)));
typedef _Float16 v8h  __attribute__((ext_vector_type(8)));
typedef float    v8f  __attribute__((ext_vector_type(8)));

// ---------------------------------------------------------------------------
// WMMA fragment loaders (wave32, 16x16x32 f16)
// A-matrix layout (ISA 7.12.2): lane group g: elems 0..7 -> K=8g+e,
//                               elems 8..15 -> K=16+8g+(e-8)
// B-matrix layout: same as A: elems 0..7 -> K=8g+e, elems 8..15 -> K=16+8g+(e-8)
// row pointers must be 16B aligned.
// ---------------------------------------------------------------------------
__device__ __forceinline__ v16h fragA(const _Float16* row, int kbase, int g) {
    v8h lo = *(const v8h*)(row + kbase + 8 * g);
    v8h hi = *(const v8h*)(row + kbase + 16 + 8 * g);
    v16h r;
#pragma unroll
    for (int i = 0; i < 8; i++) { r[i] = lo[i]; r[i + 8] = hi[i]; }
    return r;
}
__device__ __forceinline__ v16h fragB(const _Float16* row, int kbase, int g) {
    v8h lo = *(const v8h*)(row + kbase + 8 * g);
    v8h hi = *(const v8h*)(row + kbase + 16 + 8 * g);
    v16h r;
#pragma unroll
    for (int i = 0; i < 8; i++) { r[i] = lo[i]; r[i + 8] = hi[i]; }
    return r;
}
__device__ __forceinline__ v8f wmma32(v16h a, v16h b, v8f c) {
    return __builtin_amdgcn_wmma_f32_16x16x32_f16(false, a, false, b, (short)0, c,
                                                  false, false);
}

// ---------------------------------------------------------------------------
// Generic WMMA GEMM: C[M,N] = A[M,K] (f32,row-major) @ B[K,N] (f32,row-major)
// EPI: 0=none 1=+bias 2=+bias,GELU(exact) 3=+bias,+residual
// Block: 256 threads = 8 waves in a 2x4 grid; each wave computes a 2x2 grid of
// 16x16 tiles (fragment reuse: 4 WMMAs per 2 A-frags + 2 B-frags).
// Block tile: 64(M) x 128(N); K staged 32 at a time in LDS as f16.
// ---------------------------------------------------------------------------
template <int EPI>
__global__ __launch_bounds__(256) void k_gemm(const float* __restrict__ A,
                                              const float* __restrict__ B,
                                              const float* __restrict__ bias,
                                              const float* __restrict__ res,
                                              float* __restrict__ C,
                                              int M, int N, int K) {
    __shared__ _Float16 Ash[64 * 40];    // [row][k] (pad 40)
    __shared__ _Float16 Bsh[128 * 40];   // transposed: [n][k]
    const int t = threadIdx.x;
    const int wave = t >> 5, lane = t & 31, g = lane >> 4, m = lane & 15;
    const int wr = wave >> 2, wc = wave & 3;     // 2 x 4 wave grid
    const int m0 = blockIdx.y * 64, n0 = blockIdx.x * 128;

    v8f acc[2][2] = {};
    for (int k0 = 0; k0 < K; k0 += 32) {
        // stage A: 64x32
#pragma unroll
        for (int i = 0; i < 8; i++) {
            int idx = t * 8 + i;
            int r = idx >> 5, c2 = idx & 31;
            int gr = m0 + r, gk = k0 + c2;
            float v = (gr < M && gk < K) ? A[(size_t)gr * K + gk] : 0.f;
            Ash[r * 40 + c2] = (_Float16)v;
        }
        // stage B: 32x128 (store transposed)
#pragma unroll
        for (int i = 0; i < 16; i++) {
            int idx = t * 16 + i;
            int kk = idx >> 7, nn2 = idx & 127;
            int gk = k0 + kk, gn = n0 + nn2;
            float v = (gk < K && gn < N) ? B[(size_t)gk * N + gn] : 0.f;
            Bsh[nn2 * 40 + kk] = (_Float16)v;
        }
        __syncthreads();
        v16h a0 = fragA(Ash + (wr * 32 + m) * 40, 0, g);
        v16h a1 = fragA(Ash + (wr * 32 + 16 + m) * 40, 0, g);
        v16h b0 = fragB(Bsh + (wc * 32 + m) * 40, 0, g);
        v16h b1 = fragB(Bsh + (wc * 32 + 16 + m) * 40, 0, g);
        acc[0][0] = wmma32(a0, b0, acc[0][0]);
        acc[0][1] = wmma32(a0, b1, acc[0][1]);
        acc[1][0] = wmma32(a1, b0, acc[1][0]);
        acc[1][1] = wmma32(a1, b1, acc[1][1]);
        __syncthreads();
    }
#pragma unroll
    for (int ti = 0; ti < 2; ti++) {
#pragma unroll
        for (int tj = 0; tj < 2; tj++) {
            int gc = n0 + wc * 32 + tj * 16 + m;
#pragma unroll
            for (int r = 0; r < 8; r++) {
                int gr = m0 + wr * 32 + ti * 16 + 8 * g + r;
                if (gr < M && gc < N) {
                    float v = acc[ti][tj][r];
                    if (EPI >= 1) v += bias[gc];
                    if (EPI == 2) v = 0.5f * v * (1.f + erff(v * 0.70710678118654752f));
                    if (EPI == 3) v += res[(size_t)gr * N + gc];
                    C[(size_t)gr * N + gc] = v;
                }
            }
        }
    }
}

// ---------------------------------------------------------------------------
// LayerNorm over rows of 512
// ---------------------------------------------------------------------------
__global__ __launch_bounds__(128) void k_layernorm(const float* __restrict__ in,
                                                   const float* __restrict__ s,
                                                   const float* __restrict__ bp,
                                                   float* __restrict__ out) {
    const int row = blockIdx.x, t = threadIdx.x;
    const float* rp = in + (size_t)row * DIMD;
    float v[4], sum = 0.f, sq = 0.f;
#pragma unroll
    for (int i = 0; i < 4; i++) {
        v[i] = rp[t + 128 * i];
        sum += v[i];
        sq += v[i] * v[i];
    }
    __shared__ float r1[128], r2[128];
    r1[t] = sum; r2[t] = sq;
    __syncthreads();
    for (int o = 64; o > 0; o >>= 1) {
        if (t < o) { r1[t] += r1[t + o]; r2[t] += r2[t + o]; }
        __syncthreads();
    }
    float mu = r1[0] * (1.f / DIMD);
    float var = r2[0] * (1.f / DIMD) - mu * mu;
    float inv = rsqrtf(var + 1e-5f);
#pragma unroll
    for (int i = 0; i < 4; i++) {
        int c = t + 128 * i;
        out[(size_t)row * DIMD + c] = (v[i] - mu) * inv * s[c] + bp[c];
    }
}

// ---------------------------------------------------------------------------
// skipcat: out[b,o,:] = sum_c w[o,c,0]*x[b,c,:] + w[o,c,1]*prev[b,c,:] + bias[o]
// ---------------------------------------------------------------------------
__global__ __launch_bounds__(128) void k_skipcat(const float* __restrict__ x,
                                                 const float* __restrict__ prev,
                                                 const float* __restrict__ w,
                                                 const float* __restrict__ bias,
                                                 float* __restrict__ out) {
    const int bo = blockIdx.x;
    const int b = bo / NN, o = bo % NN;
    const int t = threadIdx.x;
    const float* xb = x + (size_t)b * NN * DIMD;
    const float* pb = prev + (size_t)b * NN * DIMD;
    float acc[4] = {0.f, 0.f, 0.f, 0.f};
    for (int c = 0; c < NN; c++) {
        float w0 = w[((size_t)o * NN + c) * 2 + 0];
        float w1 = w[((size_t)o * NN + c) * 2 + 1];
#pragma unroll
        for (int i = 0; i < 4; i++) {
            int d = t + 128 * i;
            acc[i] += w0 * xb[(size_t)c * DIMD + d] + w1 * pb[(size_t)c * DIMD + d];
        }
    }
    float bv = bias[o];
#pragma unroll
    for (int i = 0; i < 4; i++)
        out[((size_t)b * NN + o) * DIMD + t + 128 * i] = acc[i] + bv;
}

// ---------------------------------------------------------------------------
// MCIE kernels. concat([lidar]*N) makes the begin 1x1 conv collapse to
// rowsum(W)[c] * lidar + bias.
// ---------------------------------------------------------------------------
__global__ __launch_bounds__(128) void k_begin(const float* __restrict__ lidar,
                                               const float* __restrict__ bw,
                                               const float* __restrict__ bb,
                                               float* __restrict__ x1) {
    const int b = blockIdx.x, t = threadIdx.x;
    __shared__ float S[MIDC];
    if (t < MIDC) {
        float s = 0.f;
        for (int i = 0; i < NN; i++) s += bw[t * NN + i];
        S[t] = s;
    }
    __syncthreads();
    for (int idx = t; idx < MIDC * PP; idx += 128) {
        int c = idx / PP, p = idx % PP;
        float v = S[c] * lidar[b * PP + p] + bb[c];
        x1[((size_t)b * MIDC + c) * PP + p] = v > 0.f ? v : 0.2f * v;
    }
}

// 3->3 conv (ksize 1/3/5), lrelu, GroupNorm(1) over (3,11,11)
__global__ __launch_bounds__(128) void k_mcie_block(const float* __restrict__ x1,
                                                    float* __restrict__ Y,
                                                    const float* __restrict__ w,
                                                    const float* __restrict__ bias,
                                                    const float* __restrict__ gns,
                                                    const float* __restrict__ gnb,
                                                    int xOff, int outOff, int prevOff,
                                                    int ks) {
    const int b = blockIdx.x, t = threadIdx.x;
    __shared__ float tin[3 * PP];
    __shared__ float tout[3 * PP];
    for (int idx = t; idx < 3 * PP; idx += 128) {
        int c = idx / PP, p = idx % PP;
        float v = x1[((size_t)b * MIDC + xOff + c) * PP + p];
        if (prevOff >= 0) v += Y[((size_t)b * MIDC + prevOff + c) * PP + p];
        tin[idx] = v;
    }
    __syncthreads();
    const int pad = ks / 2;
    for (int idx = t; idx < 3 * PP; idx += 128) {
        int c = idx / PP, p = idx % PP;
        int hh = p / HWW, ww = p % HWW;
        float acc = bias[c];
        for (int i = 0; i < 3; i++)
            for (int dh = 0; dh < ks; dh++)
                for (int dw = 0; dw < ks; dw++) {
                    int h2 = hh + dh - pad, w2 = ww + dw - pad;
                    if (h2 >= 0 && h2 < HWW && w2 >= 0 && w2 < HWW)
                        acc += w[((c * 3 + i) * ks + dh) * ks + dw] *
                               tin[i * PP + h2 * HWW + w2];
                }
        tout[idx] = acc > 0.f ? acc : 0.2f * acc;
    }
    __syncthreads();
    __shared__ float r1[128], r2[128];
    float s = 0.f, q = 0.f;
    for (int idx = t; idx < 3 * PP; idx += 128) {
        float v = tout[idx]; s += v; q += v * v;
    }
    r1[t] = s; r2[t] = q;
    __syncthreads();
    for (int o = 64; o > 0; o >>= 1) {
        if (t < o) { r1[t] += r1[t + o]; r2[t] += r2[t + o]; }
        __syncthreads();
    }
    float mu = r1[0] / 363.f;
    float var = r2[0] / 363.f - mu * mu;
    float inv = rsqrtf(var + 1e-5f);
    for (int idx = t; idx < 3 * PP; idx += 128) {
        int c = idx / PP, p = idx % PP;
        Y[((size_t)b * MIDC + outOff + c) * PP + p] = (tout[idx] - mu) * inv * gns[c] + gnb[c];
    }
}

// fuse 9->201 1x1 conv + lrelu -> [B*N, 121]
__global__ __launch_bounds__(128) void k_fuse(const float* __restrict__ Y,
                                              const float* __restrict__ fw,
                                              const float* __restrict__ fb,
                                              float* __restrict__ out) {
    const int bn = blockIdx.x;
    const int b = bn / NN, n = bn % NN;
    const int t = threadIdx.x;
    if (t < PP) {
        float acc = fb[n];
#pragma unroll
        for (int c = 0; c < MIDC; c++)
            acc += fw[n * MIDC + c] * Y[((size_t)b * MIDC + c) * PP + t];
        out[(size_t)bn * PP + t] = acc > 0.f ? acc : 0.2f * acc;
    }
}

// ---------------------------------------------------------------------------
// Fused attention: per (b,h,16 query rows):
//   lm = softmax(lidq @ lidk^T * s); dots = q @ k^T * s
//   attn = softmax(cw0*dots + cw1*lm + cb); o = attn @ v; merged = o @ wm + bm
// Dynamic LDS layout (60928 B):
//   [0,29952)      KV: k/lid rows [208][72] f16  |  phase3: V^T [64][224] f16
//   [29952,32256)  Qs: [16][72] f16
//   [32256,46592)  Lm: [16][224] f32             |  later: attn probs [16][224] f16
//   [46592,60928)  Dt: [16][224] f32             |  later: WmT [64][64] f16 + Os [16][64] f16
// ---------------------------------------------------------------------------
#define ATTN_SMEM 60928

__global__ __launch_bounds__(256) void k_attn(const float* __restrict__ qkv,
                                              const float* __restrict__ lidn,
                                              const float* __restrict__ wm,
                                              const float* __restrict__ bm,
                                              const float* __restrict__ cw,
                                              const float* __restrict__ cb,
                                              float* __restrict__ out) {
    extern __shared__ char smem[];
    _Float16* KV  = (_Float16*)smem;
    _Float16* Qs  = (_Float16*)(smem + 29952);
    float*    Lm  = (float*)(smem + 32256);
    float*    Dt  = (float*)(smem + 46592);
    _Float16* At  = (_Float16*)(smem + 32256);          // overlays Lm
    _Float16* WmT = (_Float16*)(smem + 46592);          // overlays Dt
    _Float16* Os  = (_Float16*)(smem + 46592 + 8192);   // after WmT

    const int t = threadIdx.x;
    const int wave = t >> 5, lane = t & 31, g = lane >> 4, m = lane & 15;
    const int b = blockIdx.y >> 3, h = blockIdx.y & 7;
    const int q0 = blockIdx.x * 16;
    const float c0 = cw[0], c1 = cw[1], cbv = cb[0];

    // ---------------- Phase 1: lidar matrix ----------------
    for (int idx = t; idx < 16 * DH; idx += 256) {
        int r = idx >> 6, d = idx & 63, n = q0 + r;
        float v = (n < NN) ? lidn[((size_t)b * NN + n) * DIMD + h * DH + d] : 0.f;
        Qs[r * 72 + d] = (_Float16)v;
    }
    for (int idx = t; idx < NP208 * DH; idx += 256) {
        int r = idx >> 6, d = idx & 63;
        float v = (r < NN) ? lidn[((size_t)b * NN + r) * DIMD + h * DH + d] : 0.f;
        KV[r * 72 + d] = (_Float16)v;
    }
    __syncthreads();
    for (int ct = wave; ct < NTK; ct += 8) {
        v8f acc = {};
#pragma unroll
        for (int kstep = 0; kstep < 2; kstep++) {
            v16h a = fragA(Qs + m * 72, kstep * 32, g);
            v16h bf = fragB(KV + (ct * 16 + m) * 72, kstep * 32, g);
            acc = wmma32(a, bf, acc);
        }
#pragma unroll
        for (int r = 0; r < 8; r++)
            Lm[(8 * g + r) * NP224 + ct * 16 + m] = acc[r] * SCALEF;
    }
    __syncthreads();
    {   // softmax(Lm) row-wise, in place (masked cols -> 0)
        int row = t >> 4, j0 = t & 15;
        float vals[14], mx = -1e30f;
#pragma unroll
        for (int k = 0; k < 14; k++) {
            int col = j0 + 16 * k;
            float v = (col < NN) ? Lm[row * NP224 + col] : -1e30f;
            vals[k] = v;
            mx = fmaxf(mx, v);
        }
        for (int o = 8; o >= 1; o >>= 1) mx = fmaxf(mx, __shfl_xor(mx, o, 16));
        float es[14], s = 0.f;
#pragma unroll
        for (int k = 0; k < 14; k++) { es[k] = expf(vals[k] - mx); s += es[k]; }
        for (int o = 8; o >= 1; o >>= 1) s += __shfl_xor(s, o, 16);
        float inv = 1.f / s;
#pragma unroll
        for (int k = 0; k < 14; k++) {
            int col = j0 + 16 * k;
            Lm[row * NP224 + col] = (col < NN) ? es[k] * inv : 0.f;
        }
    }
    __syncthreads();

    // ---------------- Phase 2: dots & mid softmax ----------------
    for (int idx = t; idx < 16 * DH; idx += 256) {
        int r = idx >> 6, d = idx & 63, n = q0 + r;
        float v = (n < NN) ? qkv[((size_t)b * NN + n) * 1536 + h * DH + d] : 0.f;
        Qs[r * 72 + d] = (_Float16)v;
    }
    for (int idx = t; idx < NP208 * DH; idx += 256) {
        int r = idx >> 6, d = idx & 63;
        float v = (r < NN) ? qkv[((size_t)b * NN + r) * 1536 + 512 + h * DH + d] : 0.f;
        KV[r * 72 + d] = (_Float16)v;
    }
    __syncthreads();
    for (int ct = wave; ct < NTK; ct += 8) {
        v8f acc = {};
#pragma unroll
        for (int kstep = 0; kstep < 2; kstep++) {
            v16h a = fragA(Qs + m * 72, kstep * 32, g);
            v16h bf = fragB(KV + (ct * 16 + m) * 72, kstep * 32, g);
            acc = wmma32(a, bf, acc);
        }
#pragma unroll
        for (int r = 0; r < 8; r++)
            Dt[(8 * g + r) * NP224 + ct * 16 + m] = acc[r] * SCALEF;
    }
    __syncthreads();
    {   // mid = c0*dots + c1*lm + cb ; softmax -> attn probs f16 (in Lm region)
        int row = t >> 4, j0 = t & 15;
        float vals[14], mx = -1e30f;
#pragma unroll
        for (int k = 0; k < 14; k++) {
            int col = j0 + 16 * k;
            float v = (col < NN)
                          ? (c0 * Dt[row * NP224 + col] + c1 * Lm[row * NP224 + col] + cbv)
                          : -1e30f;
            vals[k] = v;
            mx = fmaxf(mx, v);
        }
        for (int o = 8; o >= 1; o >>= 1) mx = fmaxf(mx, __shfl_xor(mx, o, 16));
        float es[14], s = 0.f;
#pragma unroll
        for (int k = 0; k < 14; k++) { es[k] = expf(vals[k] - mx); s += es[k]; }
        for (int o = 8; o >= 1; o >>= 1) s += __shfl_xor(s, o, 16);
        float inv = 1.f / s;
        __syncthreads();   // all Lm/Dt reads done before At overwrites Lm region
#pragma unroll
        for (int k = 0; k < 14; k++) {
            int col = j0 + 16 * k;
            At[row * NP224 + col] = (_Float16)((col < NN) ? es[k] * inv : 0.f);
        }
    }
    __syncthreads();

    // ---------------- Phase 3: stage V^T + WmT, o = attn @ v ----------------
    for (int idx = t; idx < DH * NP224; idx += 256) {
        int d = idx / NP224, ky = idx % NP224;
        float v = (ky < NN) ? qkv[((size_t)b * NN + ky) * 1536 + 1024 + h * DH + d] : 0.f;
        KV[d * NP224 + ky] = (_Float16)v;
    }
    for (int idx = t; idx < DH * DH; idx += 256) {
        int n2 = idx >> 6, k2 = idx & 63;
        WmT[n2 * DH + k2] = (_Float16)wm[k2 * DH + n2];
    }
    __syncthreads();
    if (wave < 4) {
        int ct = wave;
        v8f acc = {};
#pragma unroll
        for (int kstep = 0; kstep < 7; kstep++) {
            v16h a = fragA(At + m * NP224, kstep * 32, g);
            v16h bf = fragB(KV + (ct * 16 + m) * NP224, kstep * 32, g);
            acc = wmma32(a, bf, acc);
        }
#pragma unroll
        for (int r = 0; r < 8; r++)
            Os[(8 * g + r) * DH + ct * 16 + m] = (_Float16)acc[r];
    }
    __syncthreads();

    // ---------------- Phase 4: merged = o @ wm + bm ----------------
    if (wave < 4) {
        int ct = wave;
        v8f acc = {};
#pragma unroll
        for (int kstep = 0; kstep < 2; kstep++) {
            v16h a = fragA(Os + m * DH, kstep * 32, g);
            v16h bf = fragB(WmT + (ct * 16 + m) * DH, kstep * 32, g);
            acc = wmma32(a, bf, acc);
        }
        int dcol = ct * 16 + m;
        float bv = bm[dcol];
#pragma unroll
        for (int r = 0; r < 8; r++) {
            int n = q0 + 8 * g + r;
            if (n < NN)
                out[((size_t)b * NN + n) * DIMD + h * DH + dcol] = acc[r] + bv;
        }
    }
}

// ---------------------------------------------------------------------------
// Host launcher
// ---------------------------------------------------------------------------
extern "C" void kernel_launch(void* const* d_in, const int* in_sizes, int n_in,
                              void* d_out, int out_size, void* d_ws, size_t ws_size,
                              hipStream_t stream) {
    auto F = [&](int i) { return (const float*)d_in[i]; };
    const float* x_in = F(0);
    const float* lidar = F(1);

    // workspace carving
    size_t off = 0;
    char* wsb = (char*)d_ws;
    auto alloc = [&](size_t nf) -> float* {
        float* p = (float*)(wsb + off);
        off += nf * sizeof(float);
        return p;
    };
    float* X1m  = alloc((size_t)BB * MIDC * PP);
    float* Ym   = alloc((size_t)BB * MIDC * PP);
    float* FUSE = alloc((size_t)ROWS * PP);
    float* LID  = alloc((size_t)ROWS * DIMD);
    float* LIDN = alloc((size_t)ROWS * DIMD);
    float* XB0  = alloc((size_t)ROWS * DIMD);
    float* XB1  = alloc((size_t)ROWS * DIMD);
    float* XB2  = alloc((size_t)ROWS * DIMD);
    float* XN   = alloc((size_t)ROWS * DIMD);
    float* MRG  = alloc((size_t)ROWS * DIMD);
    float* QKV  = alloc((size_t)ROWS * 1536);
    float* HBUF = alloc((size_t)ROWS * MLPD);

    const int GM = (ROWS + 63) / 64;   // 101 row-blocks of 64

    // ---- MCIE ----
    k_begin<<<dim3(BB), dim3(128), 0, stream>>>(lidar, F(2), F(3), X1m);
    k_mcie_block<<<dim3(BB), dim3(128), 0, stream>>>(X1m, Ym, F(4), F(5), F(10), F(11),
                                                     0, 0, -1, 1);
    k_mcie_block<<<dim3(BB), dim3(128), 0, stream>>>(X1m, Ym, F(6), F(7), F(12), F(13),
                                                     3, 3, 0, 3);
    k_mcie_block<<<dim3(BB), dim3(128), 0, stream>>>(X1m, Ym, F(8), F(9), F(14), F(15),
                                                     6, 6, 3, 5);
    k_fuse<<<dim3(ROWS), dim3(128), 0, stream>>>(Ym, F(16), F(17), FUSE);
    // lid = fuse @ lidar_w + lidar_b      [6432,121]@[121,512]
    k_gemm<1><<<dim3(DIMD / 128, GM), dim3(256), 0, stream>>>(
        FUSE, F(18), F(19), nullptr, LID, ROWS, DIMD, PP);

    // x working copy
    hipMemcpyAsync(XB0, x_in, (size_t)ROWS * DIMD * sizeof(float),
                   hipMemcpyDeviceToDevice, stream);

    float* xbuf[3] = {XB0, XB1, XB2};
    float* cur = XB0;
    const float* prevp = x_in;

    for (int i = 0; i < 4; i++) {
        const int L = 20 + 15 * i;
        const float* ln1s = F(L + 0);
        const float* ln1b = F(L + 1);
        const float* wqkv = F(L + 2);
        const float* wm   = F(L + 3);
        const float* bm   = F(L + 4);
        const float* wout = F(L + 5);
        const float* bout = F(L + 6);
        const float* cw   = F(L + 7);
        const float* cb   = F(L + 8);
        const float* ln2s = F(L + 9);
        const float* ln2b = F(L + 10);
        const float* w1   = F(L + 11);
        const float* b1   = F(L + 12);
        const float* w2   = F(L + 13);
        const float* b2   = F(L + 14);

        if (i > 0) {
            const int S = 80 + 2 * (i - 1);
            float* nb = xbuf[i % 3];
            k_skipcat<<<dim3(ROWS), dim3(128), 0, stream>>>(cur, prevp, F(S), F(S + 1), nb);
            prevp = cur;
            cur = nb;
        }

        // PreNorm(Attention): same LN applied to x and lidar
        k_layernorm<<<dim3(ROWS), dim3(128), 0, stream>>>(cur, ln1s, ln1b, XN);
        k_layernorm<<<dim3(ROWS), dim3(128), 0, stream>>>(LID, ln1s, ln1b, LIDN);

        // qkv = xn @ wqkv                 [6432,512]@[512,1536]
        k_gemm<0><<<dim3(1536 / 128, GM), dim3(256), 0, stream>>>(
            XN, wqkv, nullptr, nullptr, QKV, ROWS, 1536, DIMD);

        // attention core + per-head merge
        k_attn<<<dim3(NTK, BB * HEADS), dim3(256), ATTN_SMEM, stream>>>(
            QKV, LIDN, wm, bm, cw, cb, MRG);

        // x = x + merged @ wout + bout    [6432,512]@[512,512]
        k_gemm<3><<<dim3(DIMD / 128, GM), dim3(256), 0, stream>>>(
            MRG, wout, bout, cur, cur, ROWS, DIMD, DIMD);

        // lid = ln2(lid_n)  (lid_n == LIDN; result back into LID for next layer)
        k_layernorm<<<dim3(ROWS), dim3(128), 0, stream>>>(LIDN, ln2s, ln2b, LID);

        // PreNorm(FFN)
        k_layernorm<<<dim3(ROWS), dim3(128), 0, stream>>>(cur, ln2s, ln2b, XN);
        // h = gelu(xn @ w1 + b1)          [6432,512]@[512,1024]
        k_gemm<2><<<dim3(MLPD / 128, GM), dim3(256), 0, stream>>>(
            XN, w1, b1, nullptr, HBUF, ROWS, MLPD, DIMD);
        // x = x + h @ w2 + b2             [6432,1024]@[1024,512]
        float* outC = (i == 3) ? (float*)d_out : cur;
        k_gemm<3><<<dim3(DIMD / 128, GM), dim3(256), 0, stream>>>(
            HBUF, w2, b2, cur, outC, ROWS, DIMD, MLPD);
    }
    (void)in_sizes; (void)n_in; (void)out_size; (void)ws_size;
}
